// GAT_85727547228225
// MI455X (gfx1250) — compile-verified, matrix operand order checked by reading, not run on hardware
//
#include <hip/hip_runtime.h>
#include <hip/hip_bf16.h>

typedef __attribute__((ext_vector_type(16))) __bf16 v16bf;
typedef __attribute__((ext_vector_type(8)))  float  v8f;

#define HEADS 2
#define NEG_SLOPE 0.2f

// Fragment-position helpers for 16-bit WMMA operands (ISA 7.12.2, 16x32 tile):
// relative k (0..31) + row/col (0..15) -> (lane, element slot in v16bf).
// lanes 0-15 hold K 0-7 & 16-23 (slots 0-7 & 8-15); lanes 16-31 hold K 8-15 & 24-31.
__device__ __forceinline__ int frag_lane(int kr, int rc) {
  return (((kr >> 3) & 1) << 4) | rc;
}
__device__ __forceinline__ int frag_elem(int kr) {
  return ((kr >> 1) & 8) | (kr & 7);
}

// ---------------------------------------------------------------------------
// GEMM: C[N,Ncol] = A[N,K] @ B[K,Ncol], fp32 in/out, bf16 WMMA, fp32 accum.
// Block = 128 threads = 4 waves; one 16-row M-tile x one 64-col panel per
// block. A tile (16xK) and B panel (Kx64) are staged once into LDS in
// FRAGMENT-MAJOR order ([kstep][lane][16 bf16]) so each lane's whole 32-byte
// WMMA operand is one contiguous v16bf load (2x ds_load_b128, no packing).
// Requires N%16==0 (100000 ok), Ncol%64==0 (128/64), K<=128, K%32==0.
// ---------------------------------------------------------------------------
__global__ __launch_bounds__(128)
void gemm_bf16_wmma(const float* __restrict__ A, const float* __restrict__ B,
                    float* __restrict__ C, int N, int K, int Ncol) {
  __shared__ __align__(32) __bf16 sA[16 * 128];   // [K/32][32][16]      (4 KB)
  __shared__ __align__(32) __bf16 sB[128 * 64];   // [nt][K/32][32][16] (16 KB)

  const int tid      = threadIdx.x;
  const int m0       = blockIdx.x * 16;
  const int colBlock = blockIdx.y * 64;
  const int ktiles   = K >> 5;

  // Stage A tile (16 x K): coalesced global reads, fragment-major LDS writes
  for (int i = tid; i < 16 * K; i += 128) {
    const int r = i / K, k = i - r * K;
    const int ks = k >> 5, kr = k & 31;
    sA[(((ks << 5) | frag_lane(kr, r)) << 4) | frag_elem(kr)] =
        (__bf16)A[(size_t)(m0 + r) * K + k];
  }
  // Stage B panel (K x 64): coalesced global reads, fragment-major LDS writes
  for (int i = tid; i < (K << 6); i += 128) {
    const int k = i >> 6, c = i & 63;
    const int nt = c >> 4, n = c & 15;
    const int ks = k >> 5, kr = k & 31;
    sB[((((nt * ktiles + ks) << 5) | frag_lane(kr, n)) << 4) | frag_elem(kr)] =
        (__bf16)B[(size_t)k * Ncol + colBlock + c];
  }
  __syncthreads();

  const int lane = tid & 31;
  const int wave = tid >> 5;

  const __bf16* aBase = sA;
  const __bf16* bBase = sB + ((wave * ktiles) << 9);   // * 32 lanes * 16 elems

  v8f c = {};
  for (int ks = 0; ks < ktiles; ++ks) {
    const v16bf a = *(const v16bf*)(aBase + (((ks << 5) | lane) << 4));
    const v16bf b = *(const v16bf*)(bBase + (((ks << 5) | lane) << 4));
    c = __builtin_amdgcn_wmma_f32_16x16x32_bf16(false, a, false, b,
                                                (short)0, c, false, false);
  }

  // C/D layout: VGPR i -> M = i (lanes 0-15) / i+8 (lanes 16-31), N = lane&15
  const int rbase = m0 + ((lane >> 4) << 3);
  const int col   = colBlock + (wave << 4) + (lane & 15);
#pragma unroll
  for (int i = 0; i < 8; ++i)
    C[(size_t)(rbase + i) * Ncol + col] = c[i];
}

// ---------------------------------------------------------------------------
// Per-node attention coefficients: a_s[n,h] = <h[n,h,:], att_src[h,:]>, same a_d
// ---------------------------------------------------------------------------
__global__ void node_att_kernel(const float* __restrict__ h,
                                const float* __restrict__ att_src,
                                const float* __restrict__ att_dst,
                                float* __restrict__ a_s, float* __restrict__ a_d,
                                int N, int C) {
  int idx = blockIdx.x * blockDim.x + threadIdx.x;
  if (idx >= N * HEADS) return;
  int n = idx >> 1, hh = idx & 1;
  const float* hv = h + (size_t)n * HEADS * C + (size_t)hh * C;
  const float* as = att_src + hh * C;
  const float* ad = att_dst + hh * C;
  float s = 0.f, d = 0.f;
  for (int c = 0; c < C; ++c) { float v = hv[c]; s += v * as[c]; d += v * ad[c]; }
  a_s[idx] = s;
  a_d[idx] = d;
}

__global__ void fill_kernel(float* __restrict__ p, float v, long long n) {
  long long i = (long long)blockIdx.x * blockDim.x + threadIdx.x;
  if (i < n) p[i] = v;
}

// float atomic max via signed/unsigned integer ordering idiom
__device__ __forceinline__ void atomicMaxF(float* addr, float val) {
  if (val >= 0.f) atomicMax((int*)addr, __float_as_int(val));
  else            atomicMin((unsigned int*)addr, __float_as_uint(val));
}

// ---------------------------------------------------------------------------
// Edge pass 1: e = leaky_relu(a_s[src]+a_d[dst]); store logit; segment max
// ---------------------------------------------------------------------------
__global__ void edge_logit_max_kernel(const int* __restrict__ src,
                                      const int* __restrict__ dst,
                                      long long E, long long N,
                                      const float* __restrict__ a_s,
                                      const float* __restrict__ a_d,
                                      float* __restrict__ elog,
                                      float* __restrict__ emax) {
  long long tid = (long long)blockIdx.x * blockDim.x + threadIdx.x;
  if (tid >= (E + N) * HEADS) return;
  long long e = tid >> 1;
  int hh = (int)(tid & 1);
  int s = (e < E) ? src[e] : (int)(e - E);
  int t = (e < E) ? dst[e] : (int)(e - E);
  float v = a_s[(size_t)s * HEADS + hh] + a_d[(size_t)t * HEADS + hh];
  v = (v > 0.f) ? v : NEG_SLOPE * v;
  elog[tid] = v;
  atomicMaxF(&emax[(size_t)t * HEADS + hh], v);
}

// ---------------------------------------------------------------------------
// Edge pass 2: e_exp = exp(e - emax[dst]); segment sum into denom
// ---------------------------------------------------------------------------
__global__ void edge_exp_sum_kernel(const int* __restrict__ src,
                                    const int* __restrict__ dst,
                                    long long E, long long N,
                                    float* __restrict__ eexp,
                                    const float* __restrict__ emax,
                                    float* __restrict__ denom) {
  long long tid = (long long)blockIdx.x * blockDim.x + threadIdx.x;
  if (tid >= (E + N) * HEADS) return;
  long long e = tid >> 1;
  int hh = (int)(tid & 1);
  int t = (e < E) ? dst[e] : (int)(e - E);
  float v = __expf(eexp[tid] - emax[(size_t)t * HEADS + hh]);
  eexp[tid] = v;
  atomicAdd(&denom[(size_t)t * HEADS + hh], v);
}

// ---------------------------------------------------------------------------
// Edge pass 3: accum[dst,h,:] += alpha * h[src,h,:]; 1 thread = 1 edge x 4 chans
// ---------------------------------------------------------------------------
__global__ void edge_aggregate_kernel(const int* __restrict__ src,
                                      const int* __restrict__ dst,
                                      long long E, long long N,
                                      const float* __restrict__ eexp,
                                      const float* __restrict__ denom,
                                      const float* __restrict__ hfeat,
                                      float* __restrict__ accum, int C) {
  const int HC = HEADS * C;
  const int chunks = HC >> 2;
  long long tid = (long long)blockIdx.x * blockDim.x + threadIdx.x;
  if (tid >= (E + N) * chunks) return;
  long long e = tid / chunks;
  int c4 = (int)(tid % chunks) << 2;
  int hh = c4 / C;
  int s = (e < E) ? src[e] : (int)(e - E);
  int t = (e < E) ? dst[e] : (int)(e - E);
  float alpha = eexp[e * HEADS + hh] /
                (denom[(size_t)t * HEADS + hh] + 1e-16f);
  const float4 hv = *reinterpret_cast<const float4*>(hfeat + (size_t)s * HC + c4);
  float* op = accum + (size_t)t * HC + c4;
  atomicAdd(op + 0, alpha * hv.x);
  atomicAdd(op + 1, alpha * hv.y);
  atomicAdd(op + 2, alpha * hv.z);
  atomicAdd(op + 3, alpha * hv.w);
}

// ---------------------------------------------------------------------------
// Head mean + bias (+ optional ELU)
// ---------------------------------------------------------------------------
__global__ void finalize_kernel(const float* __restrict__ accum,
                                const float* __restrict__ bias,
                                float* __restrict__ out, int N, int C,
                                int apply_elu) {
  int tid = blockIdx.x * blockDim.x + threadIdx.x;
  if (tid >= N * C) return;
  int n = tid / C, c = tid % C;
  const float* ap = accum + (size_t)n * HEADS * C;
  float m = 0.5f * (ap[c] + ap[C + c]) + bias[c];
  if (apply_elu) m = (m > 0.f) ? m : (__expf(m) - 1.f);
  out[tid] = m;
}

static inline long long cdiv_ll(long long a, long long b) { return (a + b - 1) / b; }

extern "C" void kernel_launch(void* const* d_in, const int* in_sizes, int n_in,
                              void* d_out, int out_size, void* d_ws, size_t ws_size,
                              hipStream_t stream) {
  const float* x   = (const float*)d_in[0];
  const int*   ei  = (const int*)d_in[1];
  const float* W1  = (const float*)d_in[2];
  const float* as1 = (const float*)d_in[3];
  const float* ad1 = (const float*)d_in[4];
  const float* b1  = (const float*)d_in[5];
  const float* W2  = (const float*)d_in[6];
  const float* as2 = (const float*)d_in[7];
  const float* ad2 = (const float*)d_in[8];
  const float* b2  = (const float*)d_in[9];

  const long long N    = in_sizes[0] / 128;   // Fin = 128
  const long long E    = in_sizes[1] / 2;
  const long long Etot = E + N;
  const int Fin = 128, Hid = 64, Fout = 32;

  const int* src = ei;
  const int* dst = ei + E;

  // workspace layout (floats)
  float* ws    = (float*)d_ws;
  float* hfeat = ws;                                  // N * 128  (layer feats)
  float* accum = hfeat + (size_t)N * 128;             // N * 128  (edge accum)
  float* node1 = accum + (size_t)N * 128;             // N * 64   (layer-1 out)
  float* a_s   = node1 + (size_t)N * 64;              // N * 2
  float* a_d   = a_s   + (size_t)N * 2;               // N * 2
  float* emax  = a_d   + (size_t)N * 2;               // N * 2
  float* denom = emax  + (size_t)N * 2;               // N * 2
  float* eexp  = denom + (size_t)N * 2;               // Etot * 2

  const int BT = 256;
  const long long nh    = N * HEADS;
  const long long ntotE = Etot * HEADS;

  // ================= Layer 1 (C = Hid = 64, Ncol = 128) =================
  {
    dim3 g((unsigned)(N / 16), (unsigned)((HEADS * Hid) / 64));
    gemm_bf16_wmma<<<g, 128, 0, stream>>>(x, W1, hfeat, (int)N, Fin, HEADS * Hid);
  }
  node_att_kernel<<<(unsigned)cdiv_ll(nh, BT), BT, 0, stream>>>(
      hfeat, as1, ad1, a_s, a_d, (int)N, Hid);
  fill_kernel<<<(unsigned)cdiv_ll(nh, BT), BT, 0, stream>>>(emax, -1e30f, nh);
  fill_kernel<<<(unsigned)cdiv_ll(nh, BT), BT, 0, stream>>>(denom, 0.f, nh);
  fill_kernel<<<(unsigned)cdiv_ll(N * 128, BT), BT, 0, stream>>>(accum, 0.f, N * 128);
  edge_logit_max_kernel<<<(unsigned)cdiv_ll(ntotE, BT), BT, 0, stream>>>(
      src, dst, E, N, a_s, a_d, eexp, emax);
  edge_exp_sum_kernel<<<(unsigned)cdiv_ll(ntotE, BT), BT, 0, stream>>>(
      src, dst, E, N, eexp, emax, denom);
  edge_aggregate_kernel<<<(unsigned)cdiv_ll(Etot * (HEADS * Hid / 4), BT), BT, 0, stream>>>(
      src, dst, E, N, eexp, denom, hfeat, accum, Hid);
  finalize_kernel<<<(unsigned)cdiv_ll(N * Hid, BT), BT, 0, stream>>>(
      accum, b1, node1, (int)N, Hid, /*elu=*/1);

  // ================= Layer 2 (C = Fout = 32, Ncol = 64) =================
  {
    dim3 g((unsigned)(N / 16), (unsigned)((HEADS * Fout) / 64));
    gemm_bf16_wmma<<<g, 128, 0, stream>>>(node1, W2, hfeat, (int)N, Hid, HEADS * Fout);
  }
  node_att_kernel<<<(unsigned)cdiv_ll(nh, BT), BT, 0, stream>>>(
      hfeat, as2, ad2, a_s, a_d, (int)N, Fout);
  fill_kernel<<<(unsigned)cdiv_ll(nh, BT), BT, 0, stream>>>(emax, -1e30f, nh);
  fill_kernel<<<(unsigned)cdiv_ll(nh, BT), BT, 0, stream>>>(denom, 0.f, nh);
  fill_kernel<<<(unsigned)cdiv_ll(N * HEADS * Fout, BT), BT, 0, stream>>>(
      accum, 0.f, N * HEADS * Fout);
  edge_logit_max_kernel<<<(unsigned)cdiv_ll(ntotE, BT), BT, 0, stream>>>(
      src, dst, E, N, a_s, a_d, eexp, emax);
  edge_exp_sum_kernel<<<(unsigned)cdiv_ll(ntotE, BT), BT, 0, stream>>>(
      src, dst, E, N, eexp, emax, denom);
  edge_aggregate_kernel<<<(unsigned)cdiv_ll(Etot * (HEADS * Fout / 4), BT), BT, 0, stream>>>(
      src, dst, E, N, eexp, denom, hfeat, accum, Fout);
  finalize_kernel<<<(unsigned)cdiv_ll(N * Fout, BT), BT, 0, stream>>>(
      accum, b2, (float*)d_out, (int)N, Fout, /*elu=*/0);
}
